// VariationalMonteCarlo_26955214750002
// MI455X (gfx1250) — matrix operand order repaired and checked by reading, not compiled, not run on hardware
//
#include <hip/hip_runtime.h>

typedef __attribute__((ext_vector_type(16))) _Float16 v16h;
typedef __attribute__((ext_vector_type(8)))  _Float16 v8h;
typedef __attribute__((ext_vector_type(8)))  float    v8f;

#define NH    64
#define TNH   192
#define NSITE 144
#define NS    4096
#define WAVES 8
#define SPW   16      // samples per wave
#define SPB   128     // samples per block

__device__ __forceinline__ float sigmoid_rcp(float x) {
    float e = __expf(-x);
    return __builtin_amdgcn_rcpf(1.0f + e);          // v_exp_f32 + v_rcp_f32
}
__device__ __forceinline__ float tanh_rcp(float x) {
    float e = __expf(-2.0f * x);
    return (1.0f - e) * __builtin_amdgcn_rcpf(1.0f + e);
}
__device__ __forceinline__ float rng01(unsigned a) {
    a ^= a >> 16; a *= 0x7feb352dU;
    a ^= a >> 15; a *= 0x846ca68bU;
    a ^= a >> 16;
    return (float)(a >> 8) * (1.0f / 16777216.0f);
}

__global__ __launch_bounds__(256, 1)
void vmc_gru_sampler(const float* __restrict__ W,   // [2,192]  GRU input kernel
                     const float* __restrict__ U,   // [64,192] GRU recurrent kernel
                     const float* __restrict__ B,   // [2,192]  biases (input / recurrent)
                     const float* __restrict__ Wd,  // [64,2]   dense kernel
                     const float* __restrict__ Bd,  // [2]      dense bias
                     float* __restrict__ out)       // [4096*144] samples ++ [4096] logP
{
    // U^T in f16, pre-arranged in exact WMMA A-matrix (16x32) layout.
    __shared__ __align__(32) _Float16 u_lds[12 * 2 * 32 * 16];   // 24 KB
    // Wd^T (2 rows, zero-padded to 16) in A-matrix layout, 2 K-chunks.
    __shared__ __align__(32) _Float16 wdA_lds[2 * 32 * 16];      // 2 KB
    // h state, f16, row-major [sample][unit] (per-wave region).
    __shared__ __align__(32) _Float16 h_lds[WAVES][SPW * NH];    // 16 KB
    // Fused f32 x-contribution rows (WMMA C-operand ready):
    //   row s: W[s]+b0 (+b1 for z,r columns);  row 2: x=0 variant.
    __shared__ __align__(32) float wsel_lds[3 * TNH];            // 2.25 KB
    __shared__ __align__(32) float b1n_lds[NH];                  // n-gate recurrent bias
    __shared__ int s_lds[WAVES][SPW];                            // previous samples

    const int tid  = threadIdx.x;
    const int lane = tid & 31;
    const int w    = tid >> 5;
    const int nlo  = lane & 15;      // sample column owned by this lane
    const int h2   = lane >> 4;
    const int sid  = blockIdx.x * SPB + w * SPW + nlo;

    // ---------------- one-time LDS init ----------------
    for (int idx = tid; idx < 12 * 2 * 32 * 16; idx += 256) {
        int i  = idx & 15;
        int ln = (idx >> 4) & 31;
        int jc = idx >> 9;
        int j  = jc >> 1;
        int c  = jc & 1;
        int nl = ln & 15;                                  // A-row (gate-unit) in tile
        int hh = ln >> 4;                                  // lane half -> K sub-block
        int k  = 32 * c + 16 * hh + i;                     // K (hidden-unit in)
        u_lds[idx] = (_Float16)U[k * TNH + 16 * j + nl];   // U^T[gate][k]
    }
    for (int idx = tid; idx < 2 * 32 * 16; idx += 256) {
        int i  = idx & 15;
        int ln = (idx >> 4) & 31;
        int c  = idx >> 9;
        int nl = ln & 15;
        int hh = ln >> 4;
        int k  = 32 * c + 16 * hh + i;
        wdA_lds[idx] = (nl < 2) ? (_Float16)Wd[k * 2 + nl] : (_Float16)0.0f; // Wd^T[k-row][unit]
    }
    for (int u = tid; u < TNH; u += 256) {
        float b1f = (u < 128) ? B[TNH + u] : 0.0f;   // fold b1 into z,r rows only
        float b0  = B[u];
        wsel_lds[u]           = W[u]       + b0 + b1f;
        wsel_lds[TNH + u]     = W[TNH + u] + b0 + b1f;
        wsel_lds[2 * TNH + u] = b0 + b1f;            // x = 0 (step 0)
        if (u >= 128) b1n_lds[u - 128] = B[TNH + u]; // n-gate recurrent bias
    }
    if (tid < WAVES * SPW) ((int*)s_lds)[tid] = -1;  // sentinel: x0 = 0
    __syncthreads();

    const float bd0 = Bd[0], bd1 = Bd[1];

    // Dense A-operands live in registers for the whole kernel.
    const v16h wdA0 = *(const v16h*)&wdA_lds[(0 * 32 + lane) * 16];
    const v16h wdA1 = *(const v16h*)&wdA_lds[(1 * 32 + lane) * 16];

    float hreg[4][8];   // h state, D-layout: hreg[t][g] = h[unit 16t+8*h2+g][sample nlo]
    #pragma unroll
    for (int t = 0; t < 4; ++t)
        #pragma unroll
        for (int g = 0; g < 8; ++g) hreg[t][g] = 0.0f;

    v16h hb0 = {}, hb1 = {};   // h^T B-operand, carried across iterations (h0 = 0)
    float logP = 0.0f;

    for (int step = 0; step < NSITE; ++step) {
        // previous sample for this lane's column -> uniform row select
        int sPrev  = s_lds[w][nlo];
        int rowsel = (sPrev < 0) ? 2 : sPrev;
        const float* xrow = &wsel_lds[rowsel * TNH];

        // ---- GRU: (U^T @ h^T) with bias/x fused into C; one unit-tile at a time ----
        #pragma unroll
        for (int t = 0; t < 4; ++t) {
            // C init: z,r get fused x+b0+b1 rows; n gets b1n (x_n folded in later)
            v8f az = *(const v8f*)(xrow + 16 * (t    ) + 8 * h2);
            v8f ar = *(const v8f*)(xrow + 16 * (t + 4) + 8 * h2);
            v8f an = *(const v8f*)&b1n_lds[16 * t + 8 * h2];
            v8f xn = *(const v8f*)(xrow + 16 * (t + 8) + 8 * h2);

            #pragma unroll
            for (int c = 0; c < 2; ++c) {
                const v16h hb = (c == 0) ? hb0 : hb1;
                v16h uz = *(const v16h*)&u_lds[(((t    ) * 2 + c) * 32 + lane) * 16];
                v16h ur = *(const v16h*)&u_lds[(((t + 4) * 2 + c) * 32 + lane) * 16];
                v16h un = *(const v16h*)&u_lds[(((t + 8) * 2 + c) * 32 + lane) * 16];
                az = __builtin_amdgcn_wmma_f32_16x16x32_f16(false, uz, false, hb, (short)0, az, false, false);
                ar = __builtin_amdgcn_wmma_f32_16x16x32_f16(false, ur, false, hb, (short)0, ar, false, false);
                an = __builtin_amdgcn_wmma_f32_16x16x32_f16(false, un, false, hb, (short)0, an, false, false);
            }

            v8h hv;
            #pragma unroll
            for (int g = 0; g < 8; ++g) {
                float z  = sigmoid_rcp(az[g]);                    // x,b already in acc
                float r  = sigmoid_rcp(ar[g]);
                float cd = tanh_rcp(fmaf(r, an[g], xn[g]));       // xn folded via fma
                float hn = cd + z * (hreg[t][g] - cd);
                hreg[t][g] = hn;
                hv[g] = (_Float16)hn;
            }
            // state write-back: 8 consecutive units -> one b128 store
            *(v8h*)&h_lds[w][nlo * NH + 16 * t + 8 * h2] = hv;
        }

        asm volatile("s_wait_dscnt 0" ::: "memory");
        __builtin_amdgcn_wave_barrier();

        // ---- dense head as WMMA: logits = Wd^T @ h_new^T + bd ----
        v16h nb0 = *(const v16h*)&h_lds[w][nlo * NH + 16 * h2];
        v16h nb1 = *(const v16h*)&h_lds[w][nlo * NH + 32 + 16 * h2];
        v8f dl;
        #pragma unroll
        for (int g = 0; g < 8; ++g) dl[g] = (g == 0) ? bd0 : ((g == 1) ? bd1 : 0.0f);
        dl = __builtin_amdgcn_wmma_f32_16x16x32_f16(false, wdA0, false, nb0, (short)0, dl, false, false);
        dl = __builtin_amdgcn_wmma_f32_16x16x32_f16(false, wdA1, false, nb1, (short)0, dl, false, false);

        // ---- softmax + categorical sample + logP (lanes 0..15, one sample each) ----
        if (h2 == 0) {
            float lz0 = dl[0], lz1 = dl[1];
            float mx = fmaxf(lz0, lz1);
            float e0 = __expf(lz0 - mx), e1 = __expf(lz1 - mx);
            float inv = __builtin_amdgcn_rcpf(e0 + e1);
            float p0 = e0 * inv, p1 = e1 * inv;
            float lp0 = __logf(1e-10f + p0);
            float lp1 = __logf(1e-10f + p1);
            float u = rng01((unsigned)(step * NS + sid) * 2654435761u + 0x9e3779b9u);
            int s = (u < p0) ? 0 : 1;
            logP += s ? lp1 : lp0;
            s_lds[w][nlo] = s;
            out[sid * NSITE + step] = (float)s;
        }
        asm volatile("s_wait_dscnt 0" ::: "memory");
        __builtin_amdgcn_wave_barrier();

        // fresh h^T becomes next step's GRU input (no extra loads)
        hb0 = nb0;
        hb1 = nb1;
    }

    if (h2 == 0)
        out[NS * NSITE + sid] = logP;
}

extern "C" void kernel_launch(void* const* d_in, const int* in_sizes, int n_in,
                              void* d_out, int out_size, void* d_ws, size_t ws_size,
                              hipStream_t stream) {
    // inputs: [0]=nsamples(int), [1]=W, [2]=U, [3]=b, [4]=Wd, [5]=bd
    const float* W  = (const float*)d_in[1];
    const float* U  = (const float*)d_in[2];
    const float* B  = (const float*)d_in[3];
    const float* Wd = (const float*)d_in[4];
    const float* Bd = (const float*)d_in[5];
    float* out = (float*)d_out;
    hipLaunchKernelGGL(vmc_gru_sampler, dim3(NS / SPB), dim3(256), 0, stream,
                       W, U, B, Wd, Bd, out);
}
